// HierarchicalTopology_22265110463272
// MI455X (gfx1250) — matrix operand order repaired and checked by reading, not compiled, not run on hardware
//
#include <hip/hip_runtime.h>
#include <math.h>

// ---------------------------------------------------------------------------
// MI455X (gfx1250). All matmuls via v_wmma_f32_16x16x32_bf16 (fp32 accum).
// Weights pre-converted once to bf16 transposed [N][K]; GEMM->GEMM activations
// written as packed bf16 from the epilogue. bf16 tile staging is pure copy and uses
// the CDNA5 async global->LDS path (ASYNCcnt) when the builtins exist.
// ---------------------------------------------------------------------------

typedef __attribute__((ext_vector_type(16))) __bf16 v16bf;
typedef __attribute__((ext_vector_type(8)))  float  v8f;
typedef int v4i_vs __attribute__((vector_size(16)));

union FragU { v16bf bf; unsigned u[8]; };

__device__ __forceinline__ unsigned pack_bf2(float lo, float hi) {
  union { float f; unsigned u; } a, b;
  a.f = lo; b.f = hi;
  unsigned ul = (a.u + 0x7fffu + ((a.u >> 16) & 1u)) >> 16;
  unsigned uh = (b.u + 0x7fffu + ((b.u >> 16) & 1u)) >> 16;
  return (uh << 16) | (ul & 0xffffu);
}

#if defined(__gfx1250__) && __has_builtin(__builtin_amdgcn_global_load_async_to_lds_b128)
#define HAVE_ASYNC_LDS 1
#endif

__device__ __forceinline__ void cp_b128(const unsigned* g, unsigned* l) {
#ifdef HAVE_ASYNC_LDS
  __builtin_amdgcn_global_load_async_to_lds_b128(
      (__attribute__((address_space(1))) v4i_vs*)g,
      (__attribute__((address_space(3))) v4i_vs*)l, 0, 0);
#else
  *reinterpret_cast<uint4*>(l) = *reinterpret_cast<const uint4*>(g);
#endif
}

__device__ __forceinline__ void async_wait_all() {
#ifdef HAVE_ASYNC_LDS
#if __has_builtin(__builtin_amdgcn_s_wait_asynccnt)
  __builtin_amdgcn_s_wait_asynccnt(0);
#else
  asm volatile("s_wait_asynccnt 0" ::: "memory");
#endif
#endif
}

enum { ACT_NONE = 0, ACT_GELU = 1, ACT_SIG = 2 };

template <int ACT>
__device__ __forceinline__ float activate(float x) {
  if (ACT == ACT_GELU) return 0.5f * x * (1.0f + erff(x * 0.70710678118654752f));
  if (ACT == ACT_SIG)  return 1.0f / (1.0f + expf(-x));
  return x;
}

__device__ __forceinline__ v8f wmma_bf(const FragU& a, const FragU& b, v8f c) {
  return __builtin_amdgcn_wmma_f32_16x16x32_bf16(false, a.bf, false, b.bf,
                                                 (short)0, c, false, false);
}

// ---------------------------------------------------------------------------
// Weight pre-pass: W fp32 (K,N) row-major -> WT bf16 transposed [N][K], packed
// as u32 pairs along K (row stride K/2 u32).
// ---------------------------------------------------------------------------
__global__ void wt_bf16t_k(const float* __restrict__ W, unsigned* __restrict__ WT,
                           int K, int N) {
  int i = blockIdx.x * blockDim.x + threadIdx.x;   // over N*K/2
  int p = i % (K >> 1), n = i / (K >> 1);
  float lo = W[(size_t)(2 * p) * N + n];
  float hi = W[(size_t)(2 * p + 1) * N + n];
  WT[i] = pack_bf2(lo, hi);
}

// ---------------------------------------------------------------------------
// GEMM: C[M,N] = act(A[M,K] @ Wt^T + bias (+C if ACC))
//   A: fp32 [M,K] (ABF=0, converted in staging) or packed bf16 (ABF=1)
//   Wt: bf16 transposed [N][K] packed u32, row stride ldBu (u32)
//   C: fp32 (ldC cols) or packed bf16 (OBF=1)
// Workgroup 256 thr = 8 waves; block tile 128x64; wave tile 32x32 (4 WMMAs/K).
// Double-buffered LDS, async global->LDS staging.
// ---------------------------------------------------------------------------
template <int ACT, bool ACC, bool ABF, bool OBF>
__global__ __launch_bounds__(256) void gemm_k(
    const void* __restrict__ Ain, const unsigned* __restrict__ Wt,
    const float* __restrict__ bias, void* __restrict__ Cout,
    int M, int N, int K, int ldC, int ldBu) {
  (void)M;
  __shared__ __align__(16) unsigned aT[2][128][20];
  __shared__ __align__(16) unsigned bT[2][64][20];
  const int tid = threadIdx.x;
  const int lane = tid & 31, wv = tid >> 5;
  const int wm = (wv & 3) << 5;                 // 0,32,64,96
  const int wn = (wv >> 2) << 5;                // 0,32
  const int bm = blockIdx.x << 7, bn = blockIdx.y << 6;
  const int half = lane >> 4, l15 = lane & 15;
  const int sr = tid >> 2, sp = (tid & 3) << 2; // staging row / u32-pair base

  const float*    Af = (const float*)Ain;
  const unsigned* Au = (const unsigned*)Ain;
  const int ldAu = K >> 1;

  v8f acc00 = {}, acc01 = {}, acc10 = {}, acc11 = {};

  auto stage = [&](int kk, int buf) {
    cp_b128(&Wt[(size_t)(bn + sr) * ldBu + (kk >> 1) + sp], &bT[buf][sr][sp]);
    if (ABF) {
      cp_b128(&Au[(size_t)(bm + sr) * ldAu + (kk >> 1) + sp], &aT[buf][sr][sp]);
      cp_b128(&Au[(size_t)(bm + 64 + sr) * ldAu + (kk >> 1) + sp], &aT[buf][64 + sr][sp]);
    } else {
#pragma unroll
      for (int h = 0; h < 2; ++h) {
        const float* rp = Af + (size_t)(bm + 64 * h + sr) * K + kk + (sp << 1);
        const float4 f0 = *reinterpret_cast<const float4*>(rp);
        const float4 f1 = *reinterpret_cast<const float4*>(rp + 4);
        uint4 q;
        q.x = pack_bf2(f0.x, f0.y); q.y = pack_bf2(f0.z, f0.w);
        q.z = pack_bf2(f1.x, f1.y); q.w = pack_bf2(f1.z, f1.w);
        *reinterpret_cast<uint4*>(&aT[buf][64 * h + sr][sp]) = q;
      }
    }
  };

  stage(0, 0);
  int buf = 0;
  for (int kk = 0; kk < K; kk += 32, buf ^= 1) {
    async_wait_all();
    __syncthreads();                 // tile[buf] ready; prev reads of buf^1 done
    if (kk + 32 < K) stage(kk + 32, buf ^ 1);

    FragU a0, a1, b0, b1;
#pragma unroll
    for (int v = 0; v < 8; ++v) {
      int pA = (v < 4) ? (v + (half << 2)) : (v + 4 + (half << 2));
      int pB = pA;
      a0.u[v] = aT[buf][wm + l15][pA];
      a1.u[v] = aT[buf][wm + 16 + l15][pA];
      b0.u[v] = bT[buf][wn + l15][pB];
      b1.u[v] = bT[buf][wn + 16 + l15][pB];
    }
    acc00 = wmma_bf(a0, b0, acc00);
    acc01 = wmma_bf(a0, b1, acc01);
    acc10 = wmma_bf(a1, b0, acc10);
    acc11 = wmma_bf(a1, b1, acc11);
  }

  const int col0 = bn + wn + l15;
  const float bia0 = bias ? bias[col0] : 0.0f;
  const float bia1 = bias ? bias[col0 + 16] : 0.0f;
  float*    Cf = (float*)Cout;
  unsigned* Cb = (unsigned*)Cout;
#pragma unroll
  for (int r = 0; r < 8; ++r) {
    int rowA = bm + wm + (half << 3) + r;
    int rowB = rowA + 16;
    float v00 = acc00[r] + bia0, v01 = acc01[r] + bia1;
    float v10 = acc10[r] + bia0, v11 = acc11[r] + bia1;
    if (ACC) {
      v00 += Cf[(size_t)rowA * ldC + col0];
      v01 += Cf[(size_t)rowA * ldC + col0 + 16];
      v10 += Cf[(size_t)rowB * ldC + col0];
      v11 += Cf[(size_t)rowB * ldC + col0 + 16];
    }
    v00 = activate<ACT>(v00); v01 = activate<ACT>(v01);
    v10 = activate<ACT>(v10); v11 = activate<ACT>(v11);
    if (OBF) {
      float p00 = __shfl_xor(v00, 1, 32), p01 = __shfl_xor(v01, 1, 32);
      float p10 = __shfl_xor(v10, 1, 32), p11 = __shfl_xor(v11, 1, 32);
      if (!(l15 & 1)) {
        const int ldu = ldC >> 1;
        Cb[(size_t)rowA * ldu + (col0 >> 1)]        = pack_bf2(v00, p00);
        Cb[(size_t)rowA * ldu + ((col0 + 16) >> 1)] = pack_bf2(v01, p01);
        Cb[(size_t)rowB * ldu + (col0 >> 1)]        = pack_bf2(v10, p10);
        Cb[(size_t)rowB * ldu + ((col0 + 16) >> 1)] = pack_bf2(v11, p11);
      }
    } else {
      Cf[(size_t)rowA * ldC + col0]      = v00;
      Cf[(size_t)rowA * ldC + col0 + 16] = v01;
      Cf[(size_t)rowB * ldC + col0]      = v10;
      Cf[(size_t)rowB * ldC + col0 + 16] = v11;
    }
  }
}

// ---------------------------------------------------------------------------
// Scores: both operands packed bf16 [S][512] (row stride 256 u32).
// scores[z,s,t] = dot/sqrt(512) + loc[clip(t-s,-64,63)+64], NEG if t>s.
// ---------------------------------------------------------------------------
__global__ __launch_bounds__(256) void scores_k(
    const unsigned* __restrict__ INTb, const unsigned* __restrict__ FEATb,
    const float* __restrict__ loc, float* __restrict__ scores, int S) {
  __shared__ __align__(16) unsigned aT[2][128][20];
  __shared__ __align__(16) unsigned bT[2][64][20];
  const int tid = threadIdx.x, lane = tid & 31, wv = tid >> 5;
  const int wm = (wv & 3) << 5, wn = (wv >> 2) << 5;
  const int bm = blockIdx.x << 7, bn = blockIdx.y << 6, z = blockIdx.z;
  const int half = lane >> 4, l15 = lane & 15;
  const int sr = tid >> 2, sp = (tid & 3) << 2;
  const unsigned* Ab = INTb + (size_t)z * S * 256;
  const unsigned* Fb = FEATb + (size_t)z * S * 256;

  v8f acc00 = {}, acc01 = {}, acc10 = {}, acc11 = {};

  auto stage = [&](int kk, int buf) {
    cp_b128(&Fb[(size_t)(bn + sr) * 256 + (kk >> 1) + sp], &bT[buf][sr][sp]);
    cp_b128(&Ab[(size_t)(bm + sr) * 256 + (kk >> 1) + sp], &aT[buf][sr][sp]);
    cp_b128(&Ab[(size_t)(bm + 64 + sr) * 256 + (kk >> 1) + sp], &aT[buf][64 + sr][sp]);
  };

  stage(0, 0);
  int buf = 0;
  for (int kk = 0; kk < 512; kk += 32, buf ^= 1) {
    async_wait_all();
    __syncthreads();
    if (kk + 32 < 512) stage(kk + 32, buf ^ 1);
    FragU a0, a1, b0, b1;
#pragma unroll
    for (int v = 0; v < 8; ++v) {
      int pA = (v < 4) ? (v + (half << 2)) : (v + 4 + (half << 2));
      int pB = pA;
      a0.u[v] = aT[buf][wm + l15][pA];
      a1.u[v] = aT[buf][wm + 16 + l15][pA];
      b0.u[v] = bT[buf][wn + l15][pB];
      b1.u[v] = bT[buf][wn + 16 + l15][pB];
    }
    acc00 = wmma_bf(a0, b0, acc00);
    acc01 = wmma_bf(a0, b1, acc01);
    acc10 = wmma_bf(a1, b0, acc10);
    acc11 = wmma_bf(a1, b1, acc11);
  }

  const float sc = 0.044194173824159216f;  // 1/sqrt(512)
  const int col0 = bn + wn + l15;
#pragma unroll
  for (int r = 0; r < 8; ++r) {
#pragma unroll
    for (int q = 0; q < 4; ++q) {
      int row = bm + wm + ((q >> 1) << 4) + (half << 3) + r;
      int col = col0 + ((q & 1) << 4);
      float v = (q == 0 ? acc00[r] : q == 1 ? acc01[r] : q == 2 ? acc10[r] : acc11[r]) * sc;
      int rel = col - row;
      int ib = rel < -64 ? 0 : (rel > 63 ? 127 : rel + 64);
      v += loc[ib];
      if (rel > 0) v = -1e9f;
      scores[((size_t)z * S + row) * S + col] = v;
    }
  }
}

// ---------------------------------------------------------------------------
// Top-k + softmax. 4 waves/block, one row/wave. Tie-break: lowest index.
// ---------------------------------------------------------------------------
__global__ __launch_bounds__(128) void topk_softmax_k(
    const float* __restrict__ scores, int* __restrict__ topo,
    float* __restrict__ wts, int S, int ksel) {
  extern __shared__ float rowbufs[];             // 4 * S floats
  __shared__ float tvs[4][32];
  __shared__ int   tis[4][32];
  const int lane = threadIdx.x & 31;
  const int wave = threadIdx.x >> 5;
  const int row  = blockIdx.x * 4 + wave;
  float* buf = rowbufs + (size_t)wave * S;
  const float* srow = scores + (size_t)row * S;

  for (int t = lane; t < S; t += 32) buf[t] = srow[t];
  __syncthreads();

  for (int j = 0; j < ksel; ++j) {
    float best = -INFINITY; int bi = 0x7fffffff;
    for (int t = lane; t < S; t += 32) {
      float v = buf[t];
      if (v > best || (v == best && t < bi)) { best = v; bi = t; }
    }
#pragma unroll
    for (int off = 16; off > 0; off >>= 1) {
      float ov = __shfl_xor(best, off, 32);
      int   oi = __shfl_xor(bi, off, 32);
      if (ov > best || (ov == best && oi < bi)) { best = ov; bi = oi; }
    }
    if (lane == 0) { tvs[wave][j] = best; tis[wave][j] = bi; buf[bi] = -INFINITY; }
    __syncthreads();
  }

  float mx = tvs[wave][0];
  float e = 0.0f; int myi = 0;
  if (lane < ksel) { e = expf(tvs[wave][lane] - mx); myi = tis[wave][lane]; }
  float s = e;
#pragma unroll
  for (int off = 16; off > 0; off >>= 1) s += __shfl_xor(s, off, 32);
  if (lane < ksel) {
    wts[(size_t)row * ksel + lane]  = e / s;
    topo[(size_t)row * ksel + lane] = myi;
  }
}

// ---------------------------------------------------------------------------
// Weighted gather-sum: out[b,s,:] = sum_j w[b,s,j] * msgs[b, topo[b,s,j], :]
// ---------------------------------------------------------------------------
__global__ __launch_bounds__(128) void gather_sum_k(
    const float* __restrict__ msgs, const int* __restrict__ topo,
    const float* __restrict__ wts, float* __restrict__ outb, int S, int ksel) {
  __shared__ int   sidx[32];
  __shared__ float sw[32];
  const int tid = threadIdx.x;
  const size_t tok = blockIdx.x;
  if (tid < ksel) {
    sidx[tid] = topo[tok * ksel + tid];
    sw[tid]   = wts[tok * ksel + tid];
  }
  __syncthreads();
  const float* mb = msgs + (tok / S) * (size_t)S * 512;
  const int d = tid << 2;
  float a0 = 0, a1 = 0, a2 = 0, a3 = 0;
  for (int j = 0; j < ksel; ++j) {
    const float4 mv = *reinterpret_cast<const float4*>(mb + (size_t)sidx[j] * 512 + d);
    const float w = sw[j];
    a0 += w * mv.x; a1 += w * mv.y; a2 += w * mv.z; a3 += w * mv.w;
  }
  float4 o; o.x = a0; o.y = a1; o.z = a2; o.w = a3;
  *reinterpret_cast<float4*>(outb + tok * 512 + d) = o;
}

__global__ void combine_k(const float* __restrict__ x, const float* __restrict__ a,
                          const float* __restrict__ g, float* __restrict__ o, int n) {
  int i = blockIdx.x * blockDim.x + threadIdx.x;
  if (i < n) { float gg = g[i]; o[i] = x[i] * (1.0f - gg) + a[i] * gg; }
}

__global__ void blockmean_k(const float* __restrict__ x, float* __restrict__ o,
                            int nb, int bs) {
  int i = blockIdx.x * blockDim.x + threadIdx.x;
  int d = i & 511;
  int j = (i >> 9) % nb;
  int b = i / (nb << 9);
  const float* base = x + ((size_t)b * nb * bs + (size_t)j * bs) * 512 + d;
  float sum = 0.0f;
  for (int t = 0; t < bs; ++t) sum += base[(size_t)t * 512];
  o[i] = sum * (1.0f / (float)bs);
}

__global__ void repeat_k(const float* __restrict__ src, float* __restrict__ comb,
                         int S, int bs, int colOff) {
  int i = blockIdx.x * blockDim.x + threadIdx.x;
  int d = i & 511;
  int s = (i >> 9) % S;
  int b = i / (S << 9);
  comb[((size_t)b * S + s) * 1536 + colOff + d] =
      src[((size_t)b * (S / bs) + s / bs) * 512 + d];
}

// ---------------------------------------------------------------------------
// Host orchestration
// ---------------------------------------------------------------------------
enum GMode { GM_NONE, GM_GELU, GM_SIG_ACC, GM_GELU_OBF, GM_BF16_OBF };

static void gemm(hipStream_t st, GMode m, const void* A, const unsigned* Wt,
                 const float* bias, void* C, int M, int N, int K, int ldC, int ldBu) {
  dim3 g(M / 128, N / 64), b(256);
  switch (m) {
    case GM_NONE:
      gemm_k<ACT_NONE, false, false, false><<<g, b, 0, st>>>(A, Wt, bias, C, M, N, K, ldC, ldBu); break;
    case GM_GELU:
      gemm_k<ACT_GELU, false, false, false><<<g, b, 0, st>>>(A, Wt, bias, C, M, N, K, ldC, ldBu); break;
    case GM_SIG_ACC:
      gemm_k<ACT_SIG, true, false, false><<<g, b, 0, st>>>(A, Wt, bias, C, M, N, K, ldC, ldBu); break;
    case GM_GELU_OBF:
      gemm_k<ACT_GELU, false, false, true><<<g, b, 0, st>>>(A, Wt, bias, C, M, N, K, ldC, ldBu); break;
    case GM_BF16_OBF:
      gemm_k<ACT_NONE, false, true, true><<<g, b, 0, st>>>(A, Wt, bias, C, M, N, K, ldC, ldBu); break;
  }
}

struct LevelW {
  const unsigned *i1, *i2, *f1, *f2, *msg, *agg, *gate, *out;
  const float *i1b, *i2b, *f1b, *f2b, *loc, *msgb, *aggb, *gateb, *outb;
};

static void run_level(hipStream_t st, const float* cur, int s, int ksel,
                      const LevelW& L, float* scoresB,
                      unsigned* HIDb, unsigned* INTb, unsigned* FEATb,
                      float* G2, float* G3, float* G4, float* G5,
                      int* topoI, float* topoW, float* outC, int ldOut,
                      const unsigned* outWt) {
  const int B = 2, M = B * s;
  // intents = lin(gelu(lin(cur,i1)), i2) -> bf16
  gemm(st, GM_GELU_OBF, cur, L.i1, L.i1b, HIDb, M, 256, 512, 256, 128);
  gemm(st, GM_BF16_OBF, HIDb, L.i2, L.i2b, INTb, M, 512, 256, 512, 128);
  // feats -> bf16
  gemm(st, GM_GELU_OBF, cur, L.f1, L.f1b, HIDb, M, 256, 512, 256, 128);
  gemm(st, GM_BF16_OBF, HIDb, L.f2, L.f2b, FEATb, M, 512, 256, 512, 128);
  // scores
  scores_k<<<dim3(s / 128, s / 64, B), 256, 0, st>>>(INTb, FEATb, L.loc, scoresB, s);
  // top-k + softmax
  topk_softmax_k<<<(B * s) / 4, 128, (size_t)4 * s * sizeof(float), st>>>(
      scoresB, topoI, topoW, s, ksel);
  // msgs -> G2
  gemm(st, GM_NONE, cur, L.msg, L.msgb, G2, M, 512, 512, 512, 256);
  // gather -> G3
  gather_sum_k<<<B * s, 128, 0, st>>>(G2, topoI, topoW, G3, s, ksel);
  // agg -> G4
  gemm(st, GM_NONE, G3, L.agg, L.aggb, G4, M, 512, 512, 512, 256);
  // gate = sigmoid(cur @ Wg[0:512] + agg @ Wg[512:1024] + b) -> G5
  gemm(st, GM_NONE, cur, L.gate, L.gateb, G5, M, 512, 512, 512, 512);
  gemm(st, GM_SIG_ACC, G4, L.gate + 256, nullptr, G5, M, 512, 512, 512, 512);
  // updated -> G2 (msgs dead after gather)
  combine_k<<<(M * 512) / 256, 256, 0, st>>>(cur, G4, G5, G2, M * 512);
  // processed -> outC
  gemm(st, GM_NONE, G2, outWt, L.outb, outC, M, 512, 512, ldOut, 256);
}

extern "C" void kernel_launch(void* const* d_in, const int* in_sizes, int n_in,
                              void* d_out, int out_size, void* d_ws, size_t ws_size,
                              hipStream_t stream) {
  (void)in_sizes; (void)n_in; (void)out_size; (void)ws_size;
  const int B = 2, S0 = 2048;

  // Input index map: recursive insertion-order flatten of setup_inputs():
  // x; pred[l]{i1.W,i1.b,i2.W,i2.b,f1.W,f1.b,f2.W,f2.b,loc};
  // mp[l]{msg,agg,gate,out}{W,b}; agg[0..1]{W,b}; dist[0..1]{W,b}; fusion{W,b}
  auto P = [&](int i) { return (const float*)d_in[i]; };
  const int IX_X = 0;
  auto PRED = [](int l) { return 1 + l * 9; };
  auto MP   = [](int l) { return 28 + l * 8; };
  const int IX_AGG = 52, IX_DIST = 56, IX_FUS = 60;

  // Workspace layout (4-byte units)
  float* ws = (float*)d_ws;
  const size_t SZ_COMB = (size_t)B * S0 * 1536;
  const size_t SZ_SCR  = (size_t)B * S0 * S0;
  const size_t SZ_G    = (size_t)B * S0 * 512;
  const size_t SZ_CUR  = (size_t)B * 512 * 512;
  const size_t SZ_TOP  = (size_t)B * S0 * 32;
  float* combined = ws;
  float* scoresB  = combined + SZ_COMB;
  float* G0 = scoresB + SZ_SCR;
  float* G1 = G0 + SZ_G;
  float* G2 = G1 + SZ_G;
  float* G3 = G2 + SZ_G;
  float* G4 = G3 + SZ_G;
  float* G5 = G4 + SZ_G;
  float* CUR = G5 + SZ_G;
  int*   topoI = (int*)(CUR + SZ_CUR);
  float* topoW = (float*)(topoI + SZ_TOP);
  unsigned* HIDb  = (unsigned*)(topoW + SZ_TOP);            // M*256 bf16 = 524288 u32
  unsigned* INTb  = HIDb + (size_t)B * S0 * 128;
  unsigned* FEATb = INTb + (size_t)B * S0 * 256;
  unsigned* wtArena = FEATb + (size_t)B * S0 * 256;

  const float* x = P(IX_X);

  // -------- one-time weight conversion to bf16 transposed [N][K] ----------
  size_t wtOff = 0;
  auto cvt = [&](int ix, int K, int N) -> const unsigned* {
    unsigned* dst = wtArena + wtOff;
    size_t nu = (size_t)N * K / 2;
    wtOff += nu;
    wt_bf16t_k<<<(unsigned)(nu / 256), 256, 0, stream>>>(P(ix), dst, K, N);
    return dst;
  };

  LevelW LV[3];
  for (int l = 0; l < 3; ++l) {
    int pb = PRED(l), mb = MP(l);
    LV[l].i1 = cvt(pb + 0, 512, 256);  LV[l].i1b = P(pb + 1);
    LV[l].i2 = cvt(pb + 2, 256, 512);  LV[l].i2b = P(pb + 3);
    LV[l].f1 = cvt(pb + 4, 512, 256);  LV[l].f1b = P(pb + 5);
    LV[l].f2 = cvt(pb + 6, 256, 512);  LV[l].f2b = P(pb + 7);
    LV[l].loc = P(pb + 8);
    LV[l].msg  = cvt(mb + 0, 512, 512);  LV[l].msgb  = P(mb + 1);
    LV[l].agg  = cvt(mb + 2, 512, 512);  LV[l].aggb  = P(mb + 3);
    LV[l].gate = cvt(mb + 4, 1024, 512); LV[l].gateb = P(mb + 5);
    LV[l].out  = cvt(mb + 6, 512, 512);  LV[l].outb  = P(mb + 7);
  }
  const unsigned* aggW0  = cvt(IX_AGG + 0, 512, 512);
  const unsigned* aggW1  = cvt(IX_AGG + 2, 512, 512);
  const unsigned* distW0 = cvt(IX_DIST + 0, 512, 512);
  const unsigned* distW1 = cvt(IX_DIST + 2, 512, 512);
  const unsigned* fusW   = cvt(IX_FUS + 0, 1536, 512);

  // ------------------- Level 0 (bs=1, s=2048, k=32) -------------------
  run_level(stream, x, 2048, 32, LV[0], scoresB, HIDb, INTb, FEATb,
            G2, G3, G4, G5, topoI, topoW, combined, 1536, LV[0].out);

  // ------------------- Level 1 (bs=4, nb=512, k=16) -------------------
  {
    int nb = 512, bs = 4;
    blockmean_k<<<(B * nb * 512) / 256, 256, 0, stream>>>(x, G3, nb, bs);
    gemm(stream, GM_GELU, G3, aggW0, P(IX_AGG + 1), CUR, B * nb, 512, 512, 512, 256);
    run_level(stream, CUR, nb, 16, LV[1], scoresB, HIDb, INTb, FEATb,
              G2, G3, G4, G5, topoI, topoW, G0, 512, LV[1].out);
    gemm(stream, GM_NONE, G0, distW0, P(IX_DIST + 1), G1, B * nb, 512, 512, 512, 256);
    repeat_k<<<(B * S0 * 512) / 256, 256, 0, stream>>>(G1, combined, S0, bs, 512);
  }

  // ------------------- Level 2 (bs=16, nb=128, k=8) -------------------
  {
    int nb = 128, bs = 16;
    blockmean_k<<<(B * nb * 512) / 256, 256, 0, stream>>>(x, G3, nb, bs);
    gemm(stream, GM_GELU, G3, aggW1, P(IX_AGG + 3), CUR, B * nb, 512, 512, 512, 256);
    run_level(stream, CUR, nb, 8, LV[2], scoresB, HIDb, INTb, FEATb,
              G2, G3, G4, G5, topoI, topoW, G0, 512, LV[2].out);
    gemm(stream, GM_NONE, G0, distW1, P(IX_DIST + 3), G1, B * nb, 512, 512, 512, 256);
    repeat_k<<<(B * S0 * 512) / 256, 256, 0, stream>>>(G1, combined, S0, bs, 1024);
  }

  // ------------------- Fusion: (B*S0,1536) @ (1536,512) -------------------
  gemm(stream, GM_NONE, combined, fusW, P(IX_FUS + 1), (float*)d_out,
       B * S0, 512, 1536, 512, 768);
}
